// TopKGating_1571958030852
// MI455X (gfx1250) — compile-verified, matrix operand order checked by reading, not run on hardware
//
#include <hip/hip_runtime.h>
#include <hip/hip_bf16.h>

// ---- problem constants (match reference) ----
#define DIM   4096
#define NEXP  64
#define TOPK  8

// ---- tiling ----
#define WAVES          8
#define ROWS_PER_WAVE  16
#define ROWS_PER_WG    (WAVES * ROWS_PER_WAVE)   // 128 rows / workgroup
#define KC             128                       // K chunk staged in LDS
#define WSTRIDE        136                       // padded ushort stride (4-bank step/expert)

typedef __bf16 v16bf __attribute__((ext_vector_type(16)));
typedef __bf16 v2bf  __attribute__((ext_vector_type(2)));
typedef float  v8f   __attribute__((ext_vector_type(8)));

union Frag16 { v16bf v; v2bf p[8]; unsigned short u[16]; };

// hardware packed convert: 2 floats -> 2 bf16 in one instruction
__device__ __forceinline__ v2bf pkbf(float lo, float hi) {
#if __has_builtin(__builtin_amdgcn_cvt_pk_bf16_f32)
  return __builtin_amdgcn_cvt_pk_bf16_f32(lo, hi);
#else
  v2bf r; r.x = (__bf16)lo; r.y = (__bf16)hi; return r;
#endif
}

__global__ __launch_bounds__(256, 2)
void topk_gate_kernel(const float* __restrict__ x,
                      const float* __restrict__ gw,
                      const float* __restrict__ gb,
                      float* __restrict__ out, int n)
{
  __shared__ unsigned short wlds[NEXP * WSTRIDE];     // bf16 weight chunk [64 x 128] padded
  __shared__ float slds[ROWS_PER_WG * NEXP];          // scores [128 x 64]
  __shared__ float blds[NEXP];                        // bias

  const int tid  = threadIdx.x;
  const int wave = tid >> 5;
  const int lane = tid & 31;
  const int half = lane >> 4;                         // 0: lanes 0-15, 1: lanes 16-31
  const int l16  = lane & 15;
  const int rowBase = blockIdx.x * ROWS_PER_WG;
  const int myRow   = rowBase + wave * ROWS_PER_WAVE + l16;   // A row this lane feeds
  const int kHalfOff = half ? 8 : 0;                  // ISA 16-bit A/B lane-half K offset

  if (tid < NEXP) blds[tid] = gb[tid];

  v8f acc[4];
  #pragma unroll
  for (int t = 0; t < 4; ++t)
    acc[t] = (v8f){0.f,0.f,0.f,0.f,0.f,0.f,0.f,0.f};

  // staging geometry: 8 float4 per thread, coalesced within expert rows
  const int q0  = tid;                                // base float4 index
  for (int kb = 0; kb < DIM; kb += KC) {
    __syncthreads();                                  // previous chunk reads done

    // ---- stage gate_w[0:64, kb:kb+128] -> LDS bf16
    // phase 1: issue ALL global loads (stay in flight together)
    float4 wv[8];
    #pragma unroll
    for (int i = 0; i < 8; ++i) {
      int q  = q0 + i * 256;                          // float4 index
      int e  = q >> 5;                                // 32 float4 per expert row
      int k4 = (q & 31) << 2;
      wv[i] = *(const float4*)&gw[(size_t)e * DIM + kb + k4];
    }
    // phase 2: convert + LDS store (8B aligned ds_store_b64)
    #pragma unroll
    for (int i = 0; i < 8; ++i) {
      int q  = q0 + i * 256;
      int e  = q >> 5;
      int k4 = (q & 31) << 2;
      union { v2bf p[2]; uint2 d; } st;
      st.p[0] = pkbf(wv[i].x, wv[i].y);
      st.p[1] = pkbf(wv[i].z, wv[i].w);
      *(uint2*)&wlds[e * WSTRIDE + k4] = st.d;
    }
    __syncthreads();

    #pragma unroll
    for (int kk = 0; kk < KC; kk += 32) {
      // ---- A fragment: 16x32 bf16 (ISA layout: half-lane K offset, runs +0 and +16)
      const float* xr = x + (size_t)myRow * DIM + kb + kk + kHalfOff;
      if (kb + KC < DIM)
        __builtin_prefetch(xr + KC, 0, 0);            // global_prefetch_b8 next chunk
      float4 a0 = *(const float4*)(xr);
      float4 a1 = *(const float4*)(xr + 4);
      float4 a2 = *(const float4*)(xr + 16);
      float4 a3 = *(const float4*)(xr + 20);

      // ---- load ALL 4 B fragments first (8x ds_load_b128 in flight), then WMMA x4
      Frag16 bf[4];
      #pragma unroll
      for (int t = 0; t < 4; ++t) {
        const unsigned short* wp = &wlds[(t * 16 + l16) * WSTRIDE + kk + kHalfOff];
        *(uint2*)&bf[t].u[ 0] = *(const uint2*)(wp);       // K half-run +0..3
        *(uint2*)&bf[t].u[ 4] = *(const uint2*)(wp + 4);   // +4..7
        *(uint2*)&bf[t].u[ 8] = *(const uint2*)(wp + 16);  // +16..19
        *(uint2*)&bf[t].u[12] = *(const uint2*)(wp + 20);  // +20..23
      }

      Frag16 af;
      af.p[0] = pkbf(a0.x, a0.y);  af.p[1] = pkbf(a0.z, a0.w);
      af.p[2] = pkbf(a1.x, a1.y);  af.p[3] = pkbf(a1.z, a1.w);
      af.p[4] = pkbf(a2.x, a2.y);  af.p[5] = pkbf(a2.z, a2.w);
      af.p[6] = pkbf(a3.x, a3.y);  af.p[7] = pkbf(a3.z, a3.w);

      #pragma unroll
      for (int t = 0; t < 4; ++t)
        acc[t] = __builtin_amdgcn_wmma_f32_16x16x32_bf16(
            /*neg_a=*/false, af.v, /*neg_b=*/false, bf[t].v,
            /*c_mod=*/(short)0, acc[t], /*reuse_a=*/false, /*reuse_b=*/false);
    }
  }

  // ---- scores -> LDS (C/D layout: lanes 0-15 M=v, lanes 16-31 M=v+8; N = lane%16)
  __syncthreads();
  #pragma unroll
  for (int t = 0; t < 4; ++t)
    #pragma unroll
    for (int v = 0; v < 8; ++v)
      slds[(wave * 16 + v + half * 8) * NEXP + t * 16 + l16] = acc[t][v];
  __syncthreads();

  // ---- per-row top-8 + softmax + scatter (one lane per row)
  if (lane < 16) {
    const int r  = wave * 16 + lane;
    const int gr = rowBase + r;

    float best[TOPK];
    int   bidx[TOPK];
    #pragma unroll
    for (int j = 0; j < TOPK; ++j) { best[j] = -3.4e38f; bidx[j] = 0; }

    for (int e = 0; e < NEXP; ++e) {
      float cv = slds[r * NEXP + e] + blds[e];
      int   ci = e;
      #pragma unroll
      for (int j = 0; j < TOPK; ++j) {                 // stable sorted insert (strict >)
        bool gt  = cv > best[j];
        float nv = gt ? best[j] : cv;
        int   ni = gt ? bidx[j] : ci;
        best[j]  = gt ? cv : best[j];
        bidx[j]  = gt ? ci : bidx[j];
        cv = nv; ci = ni;
      }
    }

    float m = best[0];
    float wts[TOPK];
    float sum = 0.f;
    #pragma unroll
    for (int j = 0; j < TOPK; ++j) { wts[j] = __expf(best[j] - m); sum += wts[j]; }
    float inv = 1.0f / sum;
    #pragma unroll
    for (int j = 0; j < TOPK; ++j) wts[j] *= inv;

    // dense weights [N, E] (zeros except top-k)
    float* outW = out;
    for (int e = 0; e < NEXP; ++e) {
      float w = 0.f;
      #pragma unroll
      for (int j = 0; j < TOPK; ++j) w = (bidx[j] == e) ? wts[j] : w;
      outW[(size_t)gr * NEXP + e] = w;
    }
    // top-k indices (as float, output buffer is f32) and top-k weights
    float* outI = out + (size_t)n * NEXP;
    float* outT = outI + (size_t)n * TOPK;
    #pragma unroll
    for (int j = 0; j < TOPK; ++j) {
      outI[(size_t)gr * TOPK + j] = (float)bidx[j];
      outT[(size_t)gr * TOPK + j] = wts[j];
    }
  }
}

extern "C" void kernel_launch(void* const* d_in, const int* in_sizes, int n_in,
                              void* d_out, int out_size, void* d_ws, size_t ws_size,
                              hipStream_t stream) {
  const float* x  = (const float*)d_in[0];   // [N, 4096]
  const float* gw = (const float*)d_in[1];   // [64, 4096]
  const float* gb = (const float*)d_in[2];   // [64]
  float* out = (float*)d_out;                // [N*64 | N*8 | N*8]
  int n = in_sizes[0] / DIM;                 // 16384
  int grid = n / ROWS_PER_WG;                // 128 workgroups of 256 threads (8 waves)
  topk_gate_kernel<<<grid, 256, 0, stream>>>(x, gw, gb, out, n);
}
